// MultiHeadAttention_33260226740973
// MI455X (gfx1250) — matrix operand order repaired and checked by reading, compiled, not run on hardware
//
#include <hip/hip_runtime.h>
#include <hip/hip_bf16.h>

#define S_LEN  2048
#define DMODEL 1024
#define NHEADS 16
#define DK     64

typedef __attribute__((ext_vector_type(16))) __bf16 v16bf;
typedef __attribute__((ext_vector_type(8)))  __bf16 v8bf;
typedef __attribute__((ext_vector_type(4)))  __bf16 v4bf;
typedef __attribute__((ext_vector_type(8)))  float  v8f;

union V16U { v16bf v; v8bf h[2]; };

static __device__ __forceinline__ v8f wmma_bf16(v16bf a, v16bf b, v8f c) {
  return __builtin_amdgcn_wmma_f32_16x16x32_bf16(false, a, false, b, (short)0, c,
                                                 false, false);
}

// ---- operand loaders (contiguous K order k = 16*(lane>>4) + s, wave32) -------------
// A operand 16x32 bf16: per lane, contiguous 16-K run at sub*16,
// row = lane&15.  src points at tile origin (m0,k0).
static __device__ __forceinline__ v16bf load_a_bf16(const __bf16* src, int ld, int lane) {
  const int sub = lane >> 4, lr = lane & 15;
  const __bf16* p = src + (size_t)lr * ld + sub * 16;
  V16U u;
  u.h[0] = *(const v8bf*)(p);
  u.h[1] = *(const v8bf*)(p + 8);
  return u.v;
}
// Same A layout from f32 source -> convert in-register.
static __device__ __forceinline__ v16bf load_a_f32(const float* src, int ld, int lane) {
  const int sub = lane >> 4, lr = lane & 15;
  const float* p = src + (size_t)lr * ld + sub * 16;
  float4 f0 = ((const float4*)p)[0];
  float4 f1 = ((const float4*)p)[1];
  float4 f2 = ((const float4*)(p + 8))[0];
  float4 f3 = ((const float4*)(p + 8))[1];
  float t0[8] = {f0.x, f0.y, f0.z, f0.w, f1.x, f1.y, f1.z, f1.w};
  float t1[8] = {f2.x, f2.y, f2.z, f2.w, f3.x, f3.y, f3.z, f3.w};
  V16U u;
#pragma unroll
  for (int i = 0; i < 8; ++i) u.h[0][i] = (__bf16)t0[i];
#pragma unroll
  for (int i = 0; i < 8; ++i) u.h[1][i] = (__bf16)t1[i];
  return u.v;
}
// B operand 32x16 bf16 from [n][k] storage: lane = N, contiguous 16-K run at sub*16.
static __device__ __forceinline__ v16bf load_b_bf16(const __bf16* src, int ld, int lane) {
  const int sub = lane >> 4, lr = lane & 15;
  const __bf16* p = src + (size_t)lr * ld + sub * 16;
  V16U u;
  u.h[0] = *(const v8bf*)(p);
  u.h[1] = *(const v8bf*)(p + 8);
  return u.v;
}

// ---- f32 -> bf16 bulk convert (weights) ---------------------------------
__global__ __launch_bounds__(256) void cvt_f32_bf16(const float* __restrict__ src,
                                                    __bf16* __restrict__ dst, int n) {
  int i = (blockIdx.x * 256 + threadIdx.x) * 4;
  if (i < n) {
    float4 f = *(const float4*)(src + i);
    v4bf o;
    o[0] = (__bf16)f.x; o[1] = (__bf16)f.y; o[2] = (__bf16)f.z; o[3] = (__bf16)f.w;
    *(v4bf*)(dst + i) = o;
  }
}

// ---- mask -> additive bias row ------------------------------------------
__global__ __launch_bounds__(256) void build_bias(const int* __restrict__ mask,
                                                  float* __restrict__ bias, int n) {
  int i = blockIdx.x * 256 + threadIdx.x;
  if (i < n) bias[i] = (mask[i] != 0) ? 0.f : -1e9f;
}

// ---- GEMM: Y = X * W^T + bias; one wave computes a 16x64 strip ----------
// ASRC_BF16: 0 = X f32 (in-register cvt), 1 = X bf16.
// OMODE: 0 = bf16 head-major [b,h,s,dk]; 1 = bf16 transposed [b,h,dk,s];
//        2 = f32 row-major (final output).
template <int ASRC_BF16, int OMODE>
__global__ __launch_bounds__(32) void proj_kernel(
    const float* __restrict__ Xf, const __bf16* __restrict__ Xb,
    const __bf16* __restrict__ W, const float* __restrict__ bias,
    __bf16* __restrict__ out_bf, float* __restrict__ out_f) {
  const int lane = threadIdx.x & 31;
  const int sub = lane >> 4, lr = lane & 15;
  const int m0 = blockIdx.x * 16;
  const int n0 = blockIdx.y * 64;

  v8f cc[4] = {v8f{}, v8f{}, v8f{}, v8f{}};
#pragma unroll 2
  for (int k0 = 0; k0 < DMODEL; k0 += 32) {
    v16bf a;
    if constexpr (ASRC_BF16)
      a = load_a_bf16(Xb + (size_t)m0 * DMODEL + k0, DMODEL, lane);
    else
      a = load_a_f32(Xf + (size_t)m0 * DMODEL + k0, DMODEL, lane);
#pragma unroll
    for (int t = 0; t < 4; ++t) {
      v16bf b = load_b_bf16(W + (size_t)(n0 + t * 16) * DMODEL + k0, DMODEL, lane);
      cc[t] = wmma_bf16(a, b, cc[t]);
    }
  }

#pragma unroll
  for (int t = 0; t < 4; ++t) {
    const int col = n0 + t * 16 + lr;  // C layout: N = lane
    const float bc = bias[col];
#pragma unroll
    for (int r = 0; r < 8; ++r) {
      const int m = m0 + r + sub * 8;  // C layout: M = r + 8*sub
      const float vv = cc[t][r] + bc;
      if constexpr (OMODE == 2) {
        out_f[(size_t)m * DMODEL + col] = vv;
      } else {
        const int bi = m >> 11, s = m & (S_LEN - 1);
        const int hh = col >> 6, dk = col & 63;
        if constexpr (OMODE == 0)
          out_bf[(((size_t)bi * NHEADS + hh) * S_LEN + s) * DK + dk] = (__bf16)vv;
        else
          out_bf[(((size_t)bi * NHEADS + hh) * DK + dk) * S_LEN + s] = (__bf16)vv;
      }
    }
  }
}

// ---- transposed flash attention: one wave per (b, h, 16-query block) ----
// Computes S^T = K*Q^T so softmax stats are per-lane scalars, and
// O^T = V^T * P^T so the rescale/normalize are per-lane scalars too.
// Key->row permutation makes P^T registers *be* the context B operand.
__global__ __launch_bounds__(32) void attn_kernel(
    const __bf16* __restrict__ Qh, const __bf16* __restrict__ Kh,
    const __bf16* __restrict__ Vt, const float* __restrict__ bias,
    __bf16* __restrict__ ctx) {
  const int lane = threadIdx.x & 31;
  const int sub = lane >> 4, lr = lane & 15;
  const int q0 = blockIdx.x * 16;
  const int h = blockIdx.y, bi = blockIdx.z;
  const size_t bh = (size_t)bi * NHEADS + h;

  // Q^T B-operands (K-dim = d, N = query), loaded once
  const __bf16* qbase = Qh + (bh * S_LEN + q0) * DK;
  const v16bf bq0 = load_b_bf16(qbase, DK, lane);
  const v16bf bq1 = load_b_bf16(qbase + 32, DK, lane);

  const __bf16* kbh = Kh + bh * S_LEN * DK;
  const __bf16* vbh = Vt + bh * DK * S_LEN;
  const float* brow = bias + (size_t)bi * S_LEN;

  float mrun = -1e30f, lrun = 0.f;
  v8f o0 = {}, o1 = {}, o2 = {}, o3 = {};

  // A-row -> key permutation: rows 0..7 -> +0..7, rows 8..15 -> +16..23
  const int keyoff = lr + ((lr >> 3) & 1) * 8;

  for (int kt = 0; kt < S_LEN; kt += 32) {
    // ---- S^T: two 16-key tiles (permuted), K-dim = d_k = 64 ----
    v8f s0 = {}, s1 = {};
    {
      const __bf16* k0p = kbh + (size_t)(kt + keyoff) * DK + sub * 16;
      const __bf16* k1p = k0p + 8 * DK;  // second tile: keys +8
      V16U ua;
      ua.h[0] = *(const v8bf*)(k0p);      ua.h[1] = *(const v8bf*)(k0p + 8);
      s0 = wmma_bf16(ua.v, bq0, s0);
      ua.h[0] = *(const v8bf*)(k0p + 32); ua.h[1] = *(const v8bf*)(k0p + 40);
      s0 = wmma_bf16(ua.v, bq1, s0);
      ua.h[0] = *(const v8bf*)(k1p);      ua.h[1] = *(const v8bf*)(k1p + 8);
      s1 = wmma_bf16(ua.v, bq0, s1);
      ua.h[0] = *(const v8bf*)(k1p + 32); ua.h[1] = *(const v8bf*)(k1p + 40);
      s1 = wmma_bf16(ua.v, bq1, s1);
    }

    // ---- scale + mask bias: s0[r] <-> key kt+sub*16+r, s1[r] <-> +8+r ----
    {
      const float* bp = brow + kt + sub * 16;
      float4 mb0 = ((const float4*)bp)[0];
      float4 mb1 = ((const float4*)bp)[1];
      float4 mb2 = ((const float4*)bp)[2];
      float4 mb3 = ((const float4*)bp)[3];
      float mbs0[8] = {mb0.x, mb0.y, mb0.z, mb0.w, mb1.x, mb1.y, mb1.z, mb1.w};
      float mbs1[8] = {mb2.x, mb2.y, mb2.z, mb2.w, mb3.x, mb3.y, mb3.z, mb3.w};
#pragma unroll
      for (int r = 0; r < 8; ++r) {
        s0[r] = s0[r] * 0.125f + mbs0[r];
        s1[r] = s1[r] * 0.125f + mbs1[r];
      }
    }

    // ---- online softmax, per-lane scalars (query = lane&15) ----
    float t = s0[0];
#pragma unroll
    for (int r = 1; r < 8; ++r) t = fmaxf(t, s0[r]);
#pragma unroll
    for (int r = 0; r < 8; ++r) t = fmaxf(t, s1[r]);
    t = fmaxf(t, __shfl_xor(t, 16));  // combine the two key-halves
    const float mn = fmaxf(mrun, t);
    const float al = __expf(mrun - mn);
    const bool dead = (mn < -1e20f);  // fully-masked guard
#pragma unroll
    for (int r = 0; r < 8; ++r) {
      s0[r] = dead ? 0.f : __expf(s0[r] - mn);
      s1[r] = dead ? 0.f : __expf(s1[r] - mn);
    }
    float rs = 0.f;
#pragma unroll
    for (int r = 0; r < 8; ++r) rs += s0[r] + s1[r];
    rs += __shfl_xor(rs, 16);
    lrun = lrun * al + rs;
    mrun = mn;

    // ---- rescale O^T (alpha is a plain per-lane scalar) ----
#pragma unroll
    for (int r = 0; r < 8; ++r) {
      o0[r] *= al; o1[r] *= al; o2[r] *= al; o3[r] *= al;
    }

    // ---- P^T registers are already the context B operand (K=keys) ----
    V16U up;
#pragma unroll
    for (int r = 0; r < 8; ++r) {
      up.h[0][r] = (__bf16)s0[r];
      up.h[1][r] = (__bf16)s1[r];
    }
    const v16bf pb = up.v;

    // ---- O^T += V^T * P^T, 4 d-tiles ----
    o0 = wmma_bf16(load_a_bf16(vbh + kt, S_LEN, lane), pb, o0);
    o1 = wmma_bf16(load_a_bf16(vbh + 16 * S_LEN + kt, S_LEN, lane), pb, o1);
    o2 = wmma_bf16(load_a_bf16(vbh + 32 * S_LEN + kt, S_LEN, lane), pb, o2);
    o3 = wmma_bf16(load_a_bf16(vbh + 48 * S_LEN + kt, S_LEN, lane), pb, o3);
  }

  // ---- normalize, pack 8 consecutive d per 16B store ----
  const float inv = (lrun > 0.f) ? (1.f / lrun) : 0.f;
  const int tok = q0 + lr;  // O^T: N = lane = query
  __bf16* cp = ctx + ((size_t)bi * S_LEN + tok) * DMODEL + h * DK + sub * 8;
  v8bf ov;
#pragma unroll
  for (int r = 0; r < 8; ++r) ov[r] = (__bf16)(o0[r] * inv);
  *(v8bf*)(cp) = ov;
#pragma unroll
  for (int r = 0; r < 8; ++r) ov[r] = (__bf16)(o1[r] * inv);
  *(v8bf*)(cp + 16) = ov;
#pragma unroll
  for (int r = 0; r < 8; ++r) ov[r] = (__bf16)(o2[r] * inv);
  *(v8bf*)(cp + 32) = ov;
#pragma unroll
  for (int r = 0; r < 8; ++r) ov[r] = (__bf16)(o3[r] * inv);
  *(v8bf*)(cp + 48) = ov;
}

extern "C" void kernel_launch(void* const* d_in, const int* in_sizes, int n_in,
                              void* d_out, int out_size, void* d_ws, size_t ws_size,
                              hipStream_t stream) {
  (void)in_sizes; (void)n_in; (void)out_size; (void)ws_size;
  const float* q   = (const float*)d_in[0];
  const float* k   = (const float*)d_in[1];
  const float* v   = (const float*)d_in[2];
  const int* mask  = (const int*)d_in[3];
  const float* W_q = (const float*)d_in[4];
  const float* b_q = (const float*)d_in[5];
  const float* W_k = (const float*)d_in[6];
  const float* b_k = (const float*)d_in[7];
  const float* W_v = (const float*)d_in[8];
  const float* b_v = (const float*)d_in[9];
  const float* W_o = (const float*)d_in[10];
  const float* b_o = (const float*)d_in[11];

  const int Bb = 2;
  const size_t WN = (size_t)DMODEL * DMODEL;      // 1M elems per weight
  const size_t HN = (size_t)Bb * S_LEN * DMODEL;  // 4M elems per tensor

  char* ws = (char*)d_ws;
  __bf16* Wqb = (__bf16*)ws;
  __bf16* Wkb = Wqb + WN;
  __bf16* Wvb = Wkb + WN;
  __bf16* Wob = Wvb + WN;
  __bf16* Qh  = Wob + WN;
  __bf16* Kh  = Qh + HN;
  __bf16* Vt  = Kh + HN;
  __bf16* Ctx = Vt + HN;
  float* mbias = (float*)(Ctx + HN);              // 2*2048 f32, ~40MB ws total
  float* out = (float*)d_out;

  // 1) weights f32 -> bf16, mask -> additive bias
  dim3 cblk(256), cgrd((unsigned)(WN / (256 * 4)));
  cvt_f32_bf16<<<cgrd, cblk, 0, stream>>>(W_q, Wqb, (int)WN);
  cvt_f32_bf16<<<cgrd, cblk, 0, stream>>>(W_k, Wkb, (int)WN);
  cvt_f32_bf16<<<cgrd, cblk, 0, stream>>>(W_v, Wvb, (int)WN);
  cvt_f32_bf16<<<cgrd, cblk, 0, stream>>>(W_o, Wob, (int)WN);
  build_bias<<<dim3(Bb * S_LEN / 256), cblk, 0, stream>>>(mask, mbias, Bb * S_LEN);

  // 2) Q/K/V projections (f32 activations, bf16 weights, WMMA f32 accum)
  const int M = Bb * S_LEN;  // 4096 tokens
  dim3 pblk(32), pgrd(M / 16, DMODEL / 64);
  proj_kernel<0, 0><<<pgrd, pblk, 0, stream>>>(q, nullptr, Wqb, b_q, Qh, nullptr);
  proj_kernel<0, 0><<<pgrd, pblk, 0, stream>>>(k, nullptr, Wkb, b_k, Kh, nullptr);
  proj_kernel<0, 1><<<pgrd, pblk, 0, stream>>>(v, nullptr, Wvb, b_v, Vt, nullptr);

  // 3) transposed flash attention
  dim3 ablk(32), agrd(S_LEN / 16, NHEADS, Bb);
  attn_kernel<<<agrd, ablk, 0, stream>>>(Qh, Kh, Vt, mbias, Ctx);

  // 4) output projection -> f32
  proj_kernel<1, 2><<<pgrd, pblk, 0, stream>>>(nullptr, Ctx, Wob, b_o, nullptr, out);
}
